// LSTM_58531814310241
// MI455X (gfx1250) — matrix operand order repaired and checked by reading, not compiled, not run on hardware
//
#include <hip/hip_runtime.h>
#include <stdint.h>

// ---------------------------------------------------------------------------
// 2-layer LSTM LM for MI455X (gfx1250), bf16 WMMA everywhere.
// V=32000 E=512 H=1024 L=2 B=16 T=512
// ---------------------------------------------------------------------------

typedef __attribute__((ext_vector_type(16))) __bf16 v16bf;
typedef __attribute__((ext_vector_type(8)))  float  v8f;

union BF16Frag {           // 32 bytes = 16 bf16 = one WMMA A/B fragment per lane
    v16bf v;
    uint4 q[2];
};

__device__ __forceinline__ __bf16 f2bf(float f) {
    uint32_t u = __builtin_bit_cast(uint32_t, f);
    uint32_t r = (u + 0x7FFFu + ((u >> 16) & 1u)) >> 16;   // RNE
    unsigned short s = (unsigned short)r;
    return __builtin_bit_cast(__bf16, s);
}

__device__ __forceinline__ float sigm(float x) { return 1.0f / (1.0f + __expf(-x)); }

// Async global->LDS copy of 16B per lane (gfx1250, tracked by ASYNCcnt).
// LDS dest VGPR = low 32 bits of the generic pointer (= LDS byte offset).
__device__ __forceinline__ void async_copy16(const void* g, void* lds_generic) {
    unsigned ldsOff = (unsigned)(unsigned long long)lds_generic;
    asm volatile("global_load_async_to_lds_b128 %0, %1, off"
                 :: "v"(ldsOff), "v"(g) : "memory");
}
__device__ __forceinline__ void wait_async_le1() {
    asm volatile("s_wait_asynccnt 0x1" ::: "memory");
}
__device__ __forceinline__ void wait_async_le0() {
    asm volatile("s_wait_asynccnt 0x0" ::: "memory");
}

// ---------------------------------------------------------------------------
// f32 -> bf16 bulk convert (weights)
// ---------------------------------------------------------------------------
__global__ void cvt_bf16_kernel(const float* __restrict__ in, __bf16* __restrict__ out,
                                long long n) {
    long long i = (long long)blockIdx.x * blockDim.x + threadIdx.x;
    long long stride = (long long)gridDim.x * blockDim.x;
    for (; i < n; i += stride) out[i] = f2bf(in[i]);
}

// ---------------------------------------------------------------------------
// Embedding gather -> bf16, rows r = b*T+t, E=512
// ---------------------------------------------------------------------------
__global__ void embed_kernel(const int* __restrict__ x, const float* __restrict__ emb,
                             __bf16* __restrict__ Xbf, int total) {
    int i = blockIdx.x * blockDim.x + threadIdx.x;
    if (i >= total) return;
    int r = i >> 9;          // / E
    int e = i & 511;         // % E
    int tok = x[r];
    Xbf[i] = f2bf(emb[(long long)tok * 512 + e]);
}

// ---------------------------------------------------------------------------
// init: copy h0/c0 into state buffers, zero grid barriers
// ---------------------------------------------------------------------------
__global__ void init_state_kernel(const float* __restrict__ h0, const float* __restrict__ c0,
                                  __bf16* __restrict__ hstate, float* __restrict__ cstate,
                                  int* __restrict__ bar, int n) {
    int i = blockIdx.x * blockDim.x + threadIdx.x;
    if (i < n) { hstate[i] = f2bf(h0[i]); cstate[i] = c0[i]; }
    if (i < 8) bar[i] = 0;
}

// ---------------------------------------------------------------------------
// bf16 WMMA GEMM:  out[M,N](f32) = A[M,K](bf16) @ Bw[N,K](bf16)^T + bias0 + bias1
// block = 256 threads (8 waves), block tile 256M x 64N, wave tile 32M x 64N
// (8 accumulators), K-step 32. B tile (64 cols x 32 K) staged once per block
// via async global->LDS, triple-buffered with rotated indices (no % in loop),
// one s_barrier per K-step. All 4 B fragments are loaded before the 8 WMMAs
// so LDS reads overlap WMMA issue. Streaming f32 results stored nontemporal.
// xg_mode: row r=b*T+t stored at [(t*B+b)*ldout + n].
// ---------------------------------------------------------------------------
#define BCOL_STRIDE 80   // 64B of K data padded to 80B -> conflict-free ds_load_b128
#define BBUF_BYTES (64 * BCOL_STRIDE)

__global__ __launch_bounds__(256)
void gemm_bf16_wmma(const __bf16* __restrict__ A, int lda,
                    const __bf16* __restrict__ Bw, int ldb,
                    const float* __restrict__ bias0, const float* __restrict__ bias1,
                    float* __restrict__ out, int K, int ldout,
                    int xg_mode, int Tt, int Bb) {
    __shared__ __align__(16) char Btile[3][BBUF_BYTES];

    const int tid  = threadIdx.x;
    const int lane = tid & 31;
    const int wave = tid >> 5;
    const int half = lane >> 4;        // 0/1: lane group
    const int l15  = lane & 15;
    const int m0 = blockIdx.y * 256 + wave * 32;
    const int n0 = blockIdx.x * 64;

    // ---- async producer mapping: thread -> (col 0..63, 16B chunk 0..3) ----
    const int pcol   = tid >> 2;
    const int pchunk = tid & 3;
    const __bf16* gB = Bw + (long long)(n0 + pcol) * ldb + pchunk * 8;
    char* ldsDst[3];
#pragma unroll
    for (int b = 0; b < 3; ++b)
        ldsDst[b] = &Btile[b][pcol * BCOL_STRIDE + pchunk * 16];

    // ---- consumer fragment bases (per wave/lane) ----
    // B fragment (32x16): lane col = l15, K elems = {half*8..+7} U {16+half*8..+7}
    const char* fragBase[4];
#pragma unroll
    for (int nt = 0; nt < 4; ++nt)
        fragBase[nt] = &Btile[0][(nt * 16 + l15) * BCOL_STRIDE + half * 16];

    // A fragments (16-bit A 16x32 layout): lane row = l15,
    // K elems = {koff..+7} U {16+koff..+7}, koff = half*8. Two 16-row M tiles.
    const __bf16* aptr0 = A + (long long)(m0 + l15) * lda + half * 8;
    const __bf16* aptr1 = aptr0 + (long long)16 * lda;

    v8f acc[8] = {};                   // [mt*4 + nt]
    const int nTiles = K >> 5;

    // prologue: tiles 0 and 1 in flight
    async_copy16(gB, ldsDst[0]);
    async_copy16(gB + 32, ldsDst[1]);
    const __bf16* gBpre = gB + 64;     // next tile to issue (ki+2)

    int bufOff   = 0;                  // byte offset of compute buffer (rotates)
    int issueIdx = 2;                  // next refill buffer index (rotates)

#pragma unroll 1
    for (int ki = 0; ki < nTiles; ++ki) {
        if (ki + 1 < nTiles) wait_async_le1();   // tile ki complete (in-order)
        else                 wait_async_le0();
        __syncthreads();                          // tile visible to all waves

        const int k0 = ki << 5;
        BF16Frag a0, a1;
        a0.q[0] = *(const uint4*)(aptr0 + k0);
        a0.q[1] = *(const uint4*)(aptr0 + k0 + 16);
        a1.q[0] = *(const uint4*)(aptr1 + k0);
        a1.q[1] = *(const uint4*)(aptr1 + k0 + 16);
        __builtin_prefetch(aptr0 + k0 + 128, 0, 1);         // global_prefetch_b8
        __builtin_prefetch(aptr1 + k0 + 128, 0, 1);

        // load all B fragments first so ds reads overlap WMMA issue
        BF16Frag b[4];
#pragma unroll
        for (int nt = 0; nt < 4; ++nt) {
            const char* fb = fragBase[nt] + bufOff;
            b[nt].q[0] = *(const uint4*)(fb);
            b[nt].q[1] = *(const uint4*)(fb + 32);
        }
#pragma unroll
        for (int nt = 0; nt < 4; ++nt) {
            acc[nt] = __builtin_amdgcn_wmma_f32_16x16x32_bf16(
                false, a0.v, false, b[nt].v, (short)0, acc[nt], false, false);
            acc[4 + nt] = __builtin_amdgcn_wmma_f32_16x16x32_bf16(
                false, a1.v, false, b[nt].v, (short)0, acc[4 + nt], false, false);
        }

        // refill buffer issueIdx: its last readers ran in iter ki-1, and every
        // wave passed this iteration's barrier after finishing that read.
        if (ki + 2 < nTiles) {
            async_copy16(gBpre, ldsDst[issueIdx]);
            gBpre += 32;
            issueIdx = (issueIdx == 2) ? 0 : issueIdx + 1;
        }
        bufOff = (bufOff == 2 * BBUF_BYTES) ? 0 : bufOff + BBUF_BYTES;
    }

    // C/D layout: VGPR v -> row m0 + mt*16 + v + half*8, col n0 + nt*16 + l15
    const int rowadd = half * 8;
#pragma unroll
    for (int nt = 0; nt < 4; ++nt) {
        int n = n0 + nt * 16 + l15;
        float bsum = bias0 ? bias0[n] : 0.0f;
        if (bias1) bsum += bias1[n];
#pragma unroll
        for (int mt = 0; mt < 2; ++mt) {
#pragma unroll
            for (int v = 0; v < 8; ++v) {
                int m = m0 + mt * 16 + rowadd + v;
                float val = acc[mt * 4 + nt][v] + bsum;
                long long idx;
                if (xg_mode) {
                    int b = m / Tt, t = m % Tt;          // row r = b*T + t
                    idx = (long long)(t * Bb + b) * ldout + n;
                } else {
                    idx = (long long)m * ldout + n;
                }
                __builtin_nontemporal_store(val, &out[idx]);
            }
        }
    }
}

// ---------------------------------------------------------------------------
// Persistent LSTM recurrence for one layer.
// grid = 64 blocks (16 hidden cols each), block = 128 threads (4 waves = 4 gates).
// Wave g: 16(batch) x 16(col) tile of gate g via 32 chained WMMAs over K=H.
// Gates meet in LDS; pointwise update of c (f32) / h (bf16); monotonic
// device-scope atomic barrier between timesteps.
// ---------------------------------------------------------------------------
__global__ __launch_bounds__(128)
void lstm_layer_persistent(const float* __restrict__ xg,     // [T,B,4H]
                           const __bf16* __restrict__ Whh,   // [4H,H]
                           __bf16* __restrict__ hstate,      // [B,H] bf16
                           float* __restrict__ cstate,       // [B,H] f32
                           __bf16* __restrict__ outseq,      // [(b*T+t)*H]
                           float* __restrict__ hN, float* __restrict__ cN,
                           int* __restrict__ bar, int Tt) {
    __shared__ float gbuf[4][16][16];
    const int lane = threadIdx.x & 31;
    const int gate = threadIdx.x >> 5;
    const int half = lane >> 4;
    const int l15  = lane & 15;
    const int n0   = blockIdx.x * 16;
    const int H    = 1024;

    const __bf16* aptr = hstate + l15 * H + half * 8;                       // row = batch
    const __bf16* bptr = Whh + (long long)(gate * H + n0 + l15) * H + half * 8;

    for (int t = 0; t < Tt; ++t) {
        v8f acc = {};
#pragma unroll 1
        for (int k0 = 0; k0 < H; k0 += 32) {
            BF16Frag a, b;
            a.q[0] = *(const uint4*)(aptr + k0);
            a.q[1] = *(const uint4*)(aptr + k0 + 16);
            b.q[0] = *(const uint4*)(bptr + k0);
            b.q[1] = *(const uint4*)(bptr + k0 + 16);
            acc = __builtin_amdgcn_wmma_f32_16x16x32_bf16(
                false, a.v, false, b.v, (short)0, acc, false, false);
        }
#pragma unroll
        for (int v = 0; v < 8; ++v)
            gbuf[gate][half * 8 + v][l15] = acc[v];
        __syncthreads();

        // pointwise: 256 (m,n) cells, 128 threads -> 2 each
        for (int e = threadIdx.x; e < 256; e += 128) {
            int m = e >> 4, n = e & 15;
            int hid = n0 + n;
            long long base = (long long)(t * 16 + m) * 4096;
            float iv = sigm(gbuf[0][m][n] + __builtin_nontemporal_load(&xg[base + hid]));
            float fv = sigm(gbuf[1][m][n] + __builtin_nontemporal_load(&xg[base + 1024 + hid]));
            float gv = tanhf(gbuf[2][m][n] + __builtin_nontemporal_load(&xg[base + 2048 + hid]));
            float ov = sigm(gbuf[3][m][n] + __builtin_nontemporal_load(&xg[base + 3072 + hid]));
            float cold = cstate[m * H + hid];
            float cnew = fv * cold + iv * gv;
            float hnew = ov * tanhf(cnew);
            cstate[m * H + hid] = cnew;
            hstate[m * H + hid] = f2bf(hnew);
            outseq[((long long)m * Tt + t) * H + hid] = f2bf(hnew);
            if (t == Tt - 1) { hN[m * H + hid] = hnew; cN[m * H + hid] = cnew; }
        }
        __threadfence();
        __syncthreads();
        if (threadIdx.x == 0) {
            __hip_atomic_fetch_add(bar, 1, __ATOMIC_RELEASE, __HIP_MEMORY_SCOPE_AGENT);
            const int target = (int)gridDim.x * (t + 1);
            while (__hip_atomic_load(bar, __ATOMIC_ACQUIRE, __HIP_MEMORY_SCOPE_AGENT) < target)
                __builtin_amdgcn_s_sleep(2);
        }
        __syncthreads();
    }
}

// ---------------------------------------------------------------------------
extern "C" void kernel_launch(void* const* d_in, const int* in_sizes, int n_in,
                              void* d_out, int out_size, void* d_ws, size_t ws_size,
                              hipStream_t stream) {
    (void)in_sizes; (void)n_in; (void)out_size; (void)ws_size;
    constexpr int V = 32000, E = 512, H = 1024, B = 16, T = 512;
    constexpr int fourH = 4 * H, BT = B * T;

    const int*   x    = (const int*)  d_in[0];
    const float* h0   = (const float*)d_in[1];
    const float* c0   = (const float*)d_in[2];
    const float* emb  = (const float*)d_in[3];
    const float* Wih0 = (const float*)d_in[4];
    const float* Whh0 = (const float*)d_in[5];
    const float* bih0 = (const float*)d_in[6];
    const float* bhh0 = (const float*)d_in[7];
    const float* Wih1 = (const float*)d_in[8];
    const float* Whh1 = (const float*)d_in[9];
    const float* bih1 = (const float*)d_in[10];
    const float* bhh1 = (const float*)d_in[11];
    const float* Wfc  = (const float*)d_in[12];
    const float* bfc  = (const float*)d_in[13];
    float* outF = (float*)d_out;

    // workspace carve-up (256B aligned)
    char* base = (char*)d_ws;
    size_t off = 0;
    auto carve = [&](size_t bytes) { char* p = base + off; off = (off + bytes + 255) & ~(size_t)255; return p; };
    __bf16* Xbf    = (__bf16*)carve((size_t)BT * E * 2);
    __bf16* Wih0b  = (__bf16*)carve((size_t)fourH * E * 2);
    __bf16* Whh0b  = (__bf16*)carve((size_t)fourH * H * 2);
    __bf16* Wih1b  = (__bf16*)carve((size_t)fourH * H * 2);
    __bf16* Whh1b  = (__bf16*)carve((size_t)fourH * H * 2);
    __bf16* Wfcb   = (__bf16*)carve((size_t)V * H * 2);
    __bf16* out0b  = (__bf16*)carve((size_t)BT * H * 2);
    __bf16* out1b  = (__bf16*)carve((size_t)BT * H * 2);
    float*  xg0    = (float*) carve((size_t)T * B * fourH * 4);
    float*  xg1    = (float*) carve((size_t)T * B * fourH * 4);
    __bf16* hstate = (__bf16*)carve((size_t)2 * B * H * 2);
    float*  cstate = (float*) carve((size_t)2 * B * H * 4);
    int*    bar    = (int*)   carve(256);

    // output tail: logits | hN[L,B,H] | cN[L,B,H]
    float* hN0 = outF + (long long)BT * V;
    float* hN1 = hN0 + B * H;
    float* cN0 = hN0 + 2 * B * H;
    float* cN1 = cN0 + B * H;

    // 0) state + barrier init
    init_state_kernel<<<(2 * B * H + 255) / 256, 256, 0, stream>>>(h0, c0, hstate, cstate, bar, 2 * B * H);

    // 1) weight conversions to bf16
    cvt_bf16_kernel<<<2048, 256, 0, stream>>>(Wih0, Wih0b, (long long)fourH * E);
    cvt_bf16_kernel<<<2048, 256, 0, stream>>>(Whh0, Whh0b, (long long)fourH * H);
    cvt_bf16_kernel<<<2048, 256, 0, stream>>>(Wih1, Wih1b, (long long)fourH * H);
    cvt_bf16_kernel<<<2048, 256, 0, stream>>>(Whh1, Whh1b, (long long)fourH * H);
    cvt_bf16_kernel<<<4096, 256, 0, stream>>>(Wfc,  Wfcb,  (long long)V * H);

    // 2) embedding gather -> bf16 rows r=b*T+t
    embed_kernel<<<(BT * E + 255) / 256, 256, 0, stream>>>(x, emb, Xbf, BT * E);

    // 3) xg0 = X @ Wih0^T + bih0 + bhh0   (stored [T,B,4H])
    {
        dim3 grid(fourH / 64, BT / 256);
        gemm_bf16_wmma<<<grid, 256, 0, stream>>>(Xbf, E, Wih0b, E, bih0, bhh0,
                                                 xg0, E, fourH, 1, T, B);
    }
    // 4) layer-0 recurrence (persistent)
    lstm_layer_persistent<<<64, 128, 0, stream>>>(xg0, Whh0b, hstate, cstate,
                                                  out0b, hN0, cN0, bar, T);
    // 5) xg1 = out0 @ Wih1^T + bih1 + bhh1
    {
        dim3 grid(fourH / 64, BT / 256);
        gemm_bf16_wmma<<<grid, 256, 0, stream>>>(out0b, H, Wih1b, H, bih1, bhh1,
                                                 xg1, H, fourH, 1, T, B);
    }
    // 6) layer-1 recurrence
    lstm_layer_persistent<<<64, 128, 0, stream>>>(xg1, Whh1b, hstate + B * H, cstate + B * H,
                                                  out1b, hN1, cN1, bar + 1, T);
    // 7) logits = out1 @ Wfc^T + bfc   (537 GFLOP, dominant)
    {
        dim3 grid(V / 64, BT / 256);
        gemm_bf16_wmma<<<grid, 256, 0, stream>>>(out1b, H, Wfcb, H, bfc, nullptr,
                                                 outF, H, V, 0, T, B);
    }
}
